// Decoder_12867722018979
// MI455X (gfx1250) — matrix operand order repaired and checked by reading, not compiled, not run on hardware
//
#include <hip/hip_runtime.h>
#include <hip/hip_bf16.h>

typedef __bf16 bf16;
typedef __attribute__((ext_vector_type(16))) __bf16 v16bf;
typedef __attribute__((ext_vector_type(8)))  __bf16 v8bf;
typedef __attribute__((ext_vector_type(8)))  float  v8f;

// Problem constants
constexpr int Bn      = 262144;   // batch
constexpr int Hd      = 128;      // hidden dim
constexpr int MT      = 128;      // rows per workgroup
constexpr int THREADS = 256;      // 8 waves

// LDS layout (bf16 elements); strides padded to break 64-bank conflicts
constexpr int A_STRIDE  = 392;    // x1 tile: 128 x (384+8)
constexpr int W1_STRIDE = 392;    // layer-1 weights transposed: 128 x (384+8)
constexpr int X_STRIDE  = 264;    // x2 tile: 128 x (256+8)
constexpr int HL_STRIDE = 136;    // left layer-2 input: 128 x (128+8)
constexpr int W2L_STRIDE = 136;
constexpr int W2R_STRIDE = 264;

constexpr int A_ELEMS = MT * A_STRIDE;          // 50176
constexpr int W_ELEMS = 128 * W1_STRIDE;        // 50176 (max of all weight stagings)
constexpr int X_ELEMS = MT * X_STRIDE;          // 33792
constexpr size_t GRP_BYTE_OFF = (size_t)(A_ELEMS + W_ELEMS + X_ELEMS) * 2;   // 268288
constexpr size_t LDS_BYTES    = GRP_BYTE_OFF + MT * sizeof(int);             // 268800

__device__ __forceinline__ float sigm(float x) { return 1.0f / (1.0f + __expf(-x)); }

__device__ __forceinline__ void zero8(v8f (&a)[8]) {
#pragma unroll
  for (int i = 0; i < 8; ++i)
#pragma unroll
    for (int j = 0; j < 8; ++j) a[i][j] = 0.0f;
}

// Load a 16-element bf16 fragment as two 16B halves (hiOff in elements).
__device__ __forceinline__ v16bf ld_frag(const bf16* p, int hiOff) {
  v8bf lo = *(const v8bf*)p;
  v8bf hi = *(const v8bf*)(p + hiOff);
  v16bf r;
#pragma unroll
  for (int i = 0; i < 8; ++i) { r[i] = lo[i]; r[i + 8] = hi[i]; }
  return r;
}

// One wave computes a 16xK * Kx128 -> 16x128 tile (8 WMMA N-tiles), bf16 in, f32 acc.
// Ab: activations [row][k] row-major (stride aStride).
// Wb: weights transposed [n][k] (stride wStride). kTiles = K/32.
__device__ __forceinline__ void gemm_16xK(const bf16* Ab, int aStride,
                                          const bf16* Wb, int wStride,
                                          int kTiles, int wave, int lane,
                                          v8f (&acc)[8]) {
  const int row  = wave * 16 + (lane & 15);
  const int aSel = (lane >> 4) << 3;   // lanes 16-31 start at K+8 (ISA A 16x32 layout)
  const int col  = lane & 15;
  const int bSel = (lane >> 4) << 3;   // lanes 16-31 start at K+8 (same k order as A)
  const bf16* ap = Ab + row * aStride + aSel;
  for (int ks = 0; ks < kTiles; ++ks) {
    v16bf a = ld_frag(ap + ks * 32, 16);
#pragma unroll
    for (int nt = 0; nt < 8; ++nt) {
      v16bf b = ld_frag(Wb + (nt * 16 + col) * wStride + ks * 32 + bSel, 16);
      acc[nt] = __builtin_amdgcn_wmma_f32_16x16x32_bf16(
          /*neg_a=*/false, a, /*neg_b=*/false, b,
          /*c_mod=*/(short)0, acc[nt], /*reuse_a=*/false, /*reuse_b=*/false);
    }
  }
}

// Stage global f32 weights [K][128] row-major -> LDS bf16 transposed [128][K] (stride wStride).
__device__ __forceinline__ void stageW(const float* __restrict__ Wsrc, bf16* Wl,
                                       int K, int wStride, int tid) {
  const int total = K * 128;
  for (int idx = tid; idx < total; idx += THREADS) {
    const int k = idx >> 7, n = idx & 127;
    Wl[n * wStride + k] = (bf16)Wsrc[idx];
  }
}

// tanh(h+bh)*sigmoid(g+bg) -> LDS bf16 tile (N = 0..127)
__device__ __forceinline__ void act_lds(const v8f (&h)[8], const v8f (&g)[8],
                                        const float* __restrict__ bh,
                                        const float* __restrict__ bg,
                                        bf16* dst, int dStride, int wave, int lane) {
  const int col = lane & 15;
  const int mb  = wave * 16 + ((lane >> 4) << 3);   // D layout: lanes16-31 -> M+8
#pragma unroll
  for (int nt = 0; nt < 8; ++nt) {
    const int N = nt * 16 + col;
    const float bhv = bh[N], bgv = bg[N];
#pragma unroll
    for (int r = 0; r < 8; ++r) {
      float v = tanhf(h[nt][r] + bhv) * sigm(g[nt][r] + bgv);
      dst[(mb + r) * dStride + N] = (bf16)v;
    }
  }
}

// Final layer-2 epilogue with per-row group select, fused into the global store.
template <int RIGHT>
__device__ __forceinline__ void act_global(const v8f (&h)[8], const v8f (&g)[8],
                                           const float* __restrict__ bh,
                                           const float* __restrict__ bg,
                                           float* __restrict__ out, int m0,
                                           const int* grp, int wave, int lane) {
  const int col = lane & 15;
  const int mb  = wave * 16 + ((lane >> 4) << 3);
#pragma unroll
  for (int nt = 0; nt < 8; ++nt) {
    const int N = nt * 16 + col;
    const float bhv = bh[N], bgv = bg[N];
#pragma unroll
    for (int r = 0; r < 8; ++r) {
      const int M = mb + r;
      float v = tanhf(h[nt][r] + bhv) * sigm(g[nt][r] + bgv);
      const int gv = grp[M];
      if (RIGHT) {
        if (gv == 1) out[(size_t)(m0 + M) * Hd + N] = v;        // overwrite right rows
      } else {
        out[(size_t)(m0 + M) * Hd + N] = (gv == 0) ? v : 0.0f;  // left rows / zero pad
      }
    }
  }
}

__global__ __launch_bounds__(THREADS, 1)
void gatenn_decoder_kernel(
    const float* __restrict__ node_hidden, const float* __restrict__ node_context,
    const float* __restrict__ label_emb,   const float* __restrict__ left_emb,
    const float* __restrict__ lWh1, const float* __restrict__ lbh1,
    const float* __restrict__ lWg1, const float* __restrict__ lbg1,
    const float* __restrict__ lWh2, const float* __restrict__ lbh2,
    const float* __restrict__ lWg2, const float* __restrict__ lbg2,
    const float* __restrict__ rWh1, const float* __restrict__ rbh1,
    const float* __restrict__ rWg1, const float* __restrict__ rbg1,
    const float* __restrict__ rWh2, const float* __restrict__ rbh2,
    const float* __restrict__ rWg2, const float* __restrict__ rbg2,
    const int* __restrict__ group,
    float* __restrict__ out, int* __restrict__ mask_out) {
  extern __shared__ unsigned char smem_raw[];
  bf16* Al = (bf16*)smem_raw;            // x1 tile     [128][392]
  bf16* Wl = Al + A_ELEMS;               // weights     [128][<=392] (restaged per gate)
  bf16* Xl = Wl + W_ELEMS;               // x2 / h tile [128][<=264]
  int*  grp = (int*)(smem_raw + GRP_BYTE_OFF);

  const int tid  = threadIdx.x;
  const int wave = tid >> 5;
  const int lane = tid & 31;
  const int m0   = blockIdx.x * MT;

  // ---- Stage x1 = [hidden | context | label] as bf16; group + finished mask ----
  for (int idx = tid; idx < MT * Hd; idx += THREADS) {
    const int r = idx >> 7, c = idx & 127;
    const int gidx = (m0 + r) * Hd + c;
    Al[r * A_STRIDE + c]        = (bf16)node_hidden[gidx];
    Al[r * A_STRIDE + 128 + c]  = (bf16)node_context[gidx];
    Al[r * A_STRIDE + 256 + c]  = (bf16)label_emb[gidx];
  }
  if (tid < MT) {
    const int g = group[m0 + tid];
    grp[tid] = g;
    mask_out[m0 + tid] = (g == 2) ? 1 : 0;
  }
  __syncthreads();

  v8f accH[8], accG[8];

  // ======================= LEFT branch =======================
  stageW(lWh1, Wl, 384, W1_STRIDE, tid); __syncthreads();
  zero8(accH); gemm_16xK(Al, A_STRIDE, Wl, W1_STRIDE, 12, wave, lane, accH);
  __syncthreads();
  stageW(lWg1, Wl, 384, W1_STRIDE, tid); __syncthreads();
  zero8(accG); gemm_16xK(Al, A_STRIDE, Wl, W1_STRIDE, 12, wave, lane, accG);
  act_lds(accH, accG, lbh1, lbg1, Xl, HL_STRIDE, wave, lane);
  __syncthreads();

  stageW(lWh2, Wl, 128, W2L_STRIDE, tid); __syncthreads();
  zero8(accH); gemm_16xK(Xl, HL_STRIDE, Wl, W2L_STRIDE, 4, wave, lane, accH);
  __syncthreads();
  stageW(lWg2, Wl, 128, W2L_STRIDE, tid); __syncthreads();
  zero8(accG); gemm_16xK(Xl, HL_STRIDE, Wl, W2L_STRIDE, 4, wave, lane, accG);
  act_global<0>(accH, accG, lbh2, lbg2, out, m0, grp, wave, lane);
  __syncthreads();   // all Xl reads done before right branch overwrites it

  // ======================= RIGHT branch =======================
  stageW(rWh1, Wl, 384, W1_STRIDE, tid);
  // left_embedding -> x2 columns 128..255 (stride 264)
  for (int idx = tid; idx < MT * Hd; idx += THREADS) {
    const int r = idx >> 7, c = idx & 127;
    Xl[r * X_STRIDE + 128 + c] = (bf16)left_emb[(m0 + r) * Hd + c];
  }
  __syncthreads();
  zero8(accH); gemm_16xK(Al, A_STRIDE, Wl, W1_STRIDE, 12, wave, lane, accH);
  __syncthreads();
  stageW(rWg1, Wl, 384, W1_STRIDE, tid); __syncthreads();
  zero8(accG); gemm_16xK(Al, A_STRIDE, Wl, W1_STRIDE, 12, wave, lane, accG);
  act_lds(accH, accG, rbh1, rbg1, Xl, X_STRIDE, wave, lane);  // h -> x2 cols 0..127
  __syncthreads();

  stageW(rWh2, Wl, 256, W2R_STRIDE, tid); __syncthreads();
  zero8(accH); gemm_16xK(Xl, X_STRIDE, Wl, W2R_STRIDE, 8, wave, lane, accH);
  __syncthreads();
  stageW(rWg2, Wl, 256, W2R_STRIDE, tid); __syncthreads();
  zero8(accG); gemm_16xK(Xl, X_STRIDE, Wl, W2R_STRIDE, 8, wave, lane, accG);
  act_global<1>(accH, accG, rbh2, rbg2, out, m0, grp, wave, lane);
}

extern "C" void kernel_launch(void* const* d_in, const int* in_sizes, int n_in,
                              void* d_out, int out_size, void* d_ws, size_t ws_size,
                              hipStream_t stream) {
  (void)in_sizes; (void)n_in; (void)d_ws; (void)ws_size; (void)out_size;
  const float* node_hidden  = (const float*)d_in[0];
  const float* node_context = (const float*)d_in[1];
  const float* label_emb    = (const float*)d_in[2];
  const float* left_emb     = (const float*)d_in[3];
  const float* lWh1 = (const float*)d_in[4];  const float* lbh1 = (const float*)d_in[5];
  const float* lWg1 = (const float*)d_in[6];  const float* lbg1 = (const float*)d_in[7];
  const float* lWh2 = (const float*)d_in[8];  const float* lbh2 = (const float*)d_in[9];
  const float* lWg2 = (const float*)d_in[10]; const float* lbg2 = (const float*)d_in[11];
  const float* rWh1 = (const float*)d_in[12]; const float* rbh1 = (const float*)d_in[13];
  const float* rWg1 = (const float*)d_in[14]; const float* rbg1 = (const float*)d_in[15];
  const float* rWh2 = (const float*)d_in[16]; const float* rbh2 = (const float*)d_in[17];
  const float* rWg2 = (const float*)d_in[18]; const float* rbg2 = (const float*)d_in[19];
  const int*   group = (const int*)d_in[20];

  float* out  = (float*)d_out;
  int*   mask = (int*)((float*)d_out + (size_t)Bn * Hd);   // tuple output: [B*H f32][B i32]

  (void)hipFuncSetAttribute((const void*)gatenn_decoder_kernel,
                            hipFuncAttributeMaxDynamicSharedMemorySize, (int)LDS_BYTES);

  dim3 grid(Bn / MT), block(THREADS);
  gatenn_decoder_kernel<<<grid, block, LDS_BYTES, stream>>>(
      node_hidden, node_context, label_emb, left_emb,
      lWh1, lbh1, lWg1, lbg1, lWh2, lbh2, lWg2, lbg2,
      rWh1, rbh1, rWg1, rbg1, rWh2, rbh2, rWg2, rbg2,
      group, out, mask);
}
